// GraphTransformer_41240275976659
// MI455X (gfx1250) — matrix operand order repaired and checked by reading, not compiled, not run on hardware
//
#include <hip/hip_runtime.h>
#include <hip/hip_bf16.h>
#include <math.h>

// ---------------------------------------------------------------------------
// GraphTransformer attention for MI455X (gfx1250), wave32 WMMA bf16.
// Shapes: B=1, N=512, H=8, D=32, IN_DIM=128, EDGE_DIM=64, INNER=256.
// ---------------------------------------------------------------------------

typedef __attribute__((ext_vector_type(16))) __bf16 v16bf;
typedef __attribute__((ext_vector_type(8)))  float  v8f;

#define N_NODES 512
#define IN_DIM  128
#define INNER   256
#define HEADS   8
#define HD      32      // D per head
#define EDIM    64

__device__ __forceinline__ v8f wmma_bf16(v16bf a, v16bf b, v8f c) {
  // D(16x16 f32) = A(16x32 bf16) * B(32x16 bf16) + C
  return __builtin_amdgcn_wmma_f32_16x16x32_bf16(false, a, false, b,
                                                 (short)0, c, false, false);
}

// A fragment: 16x32 tile, rows row-major f32 in memory, converted to bf16.
// ISA 7.12.2 16-bit A layout: lane L holds row M=L&15; group g=L>>4.
//   VGPR i<4 : K = 2i,2i+1   (+8 if g==1)
//   VGPR i>=4: K = 16+2(i-4) (+8 if g==1)
__device__ __forceinline__ v16bf load_a_frag(const float* __restrict__ rowptr, int g) {
  v16bf a;
#pragma unroll
  for (int i = 0; i < 8; ++i) {
    const int kb = (i < 4) ? (2*i + 8*g) : (16 + 2*(i - 4) + 8*g);
    const float2 f = *(const float2*)(rowptr + kb);
    a[2*i]     = (__bf16)f.x;
    a[2*i + 1] = (__bf16)f.y;
  }
  return a;
}

// B fragment: 32x16 (K x N) from row-major f32 W with leading dim ld.
// lane L holds col N=L&15; g=L>>4: VGPR i<4 holds K = 2i,2i+1 (+8g); i>=4: K = 16+2(i-4) (+8g).
__device__ __forceinline__ v16bf load_b_frag(const float* __restrict__ W, int ld,
                                             int krow0, int col, int g) {
  v16bf b;
#pragma unroll
  for (int i = 0; i < 8; ++i) {
    const int kr = krow0 + ((i < 4) ? (2*i + 8*g) : (16 + 2*(i - 4) + 8*g));
    b[2*i]     = (__bf16)W[(size_t)kr       * ld + col];
    b[2*i + 1] = (__bf16)W[(size_t)(kr + 1) * ld + col];
  }
  return b;
}

// ---------------------------------------------------------------------------
// Kernel 1: q/k/v projections.  grid = (32 row-tiles, 3 matrices), block=128.
// Each wave computes 4 of the 16 N-tiles; K=128 -> 4 WMMA steps.
// ---------------------------------------------------------------------------
__global__ __launch_bounds__(128)
void gt_qkv_proj(const float* __restrict__ nodes,
                 const float* __restrict__ Wq, const float* __restrict__ bq,
                 const float* __restrict__ Wk, const float* __restrict__ bk,
                 const float* __restrict__ Wv, const float* __restrict__ bv,
                 float* __restrict__ qbuf, float* __restrict__ kbuf,
                 float* __restrict__ vbuf) {
  const int mt    = blockIdx.x;          // 16-row tile of nodes
  const int which = blockIdx.y;          // 0=q 1=k 2=v
  const int lane  = threadIdx.x & 31;
  const int wv    = threadIdx.x >> 5;    // wave 0..3
  const int g     = lane >> 4;
  const int m     = lane & 15;

  const float* W; const float* bias; float* dst;
  if (which == 0)      { W = Wq; bias = bq; dst = qbuf; }
  else if (which == 1) { W = Wk; bias = bk; dst = kbuf; }
  else                 { W = Wv; bias = bv; dst = vbuf; }

  const float* rowptr = nodes + (size_t)(mt*16 + m) * IN_DIM;
  v16bf afr[4];
#pragma unroll
  for (int ks = 0; ks < 4; ++ks) afr[ks] = load_a_frag(rowptr + ks*32, g);

#pragma unroll
  for (int ni = 0; ni < 4; ++ni) {
    const int nt  = wv*4 + ni;
    const int col = nt*16 + (lane & 15);
    v8f acc = {};
#pragma unroll
    for (int ks = 0; ks < 4; ++ks) {
      v16bf bf = load_b_frag(W, INNER, ks*32, col, g);
      acc = wmma_bf16(afr[ks], bf, acc);
    }
    const float bb = bias[col];
#pragma unroll
    for (int r = 0; r < 8; ++r) {
      const int row = mt*16 + r + 8*g;               // C/D layout: M = r + 8*(lane>>4)
      dst[(size_t)row * INNER + col] = acc[r] + bb;
    }
  }
}

// ---------------------------------------------------------------------------
// Kernel 2: fused edge-projection + attention.
// grid = (32 query tiles, 8 heads), block = 128 (4 waves).
// Pass 1: per j-tile, WMMA-project edges->E(256x32) in LDS, VALU q.(k+e)
//         into sim strip (16x512). Masked softmax stats (max-sub, == ref).
// Pass 2: recompute E (edges are L2-resident: 64 MB < 192 MB L2),
//         accumulate attn.(v+e), normalize, store.
// ---------------------------------------------------------------------------
__global__ __launch_bounds__(128)
void gt_attn(const float* __restrict__ edges, const int* __restrict__ adj,
             const float* __restrict__ We, const float* __restrict__ be,
             const float* __restrict__ qbuf, const float* __restrict__ kbuf,
             const float* __restrict__ vbuf, float* __restrict__ out) {
  const int it  = blockIdx.x;            // query tile, 16 rows
  const int h   = blockIdx.y;            // head
  const int i0  = it * 16;
  const int tid = threadIdx.x;
  const int lane = tid & 31;
  const int wv   = tid >> 5;
  const int g    = lane >> 4;

  __shared__ float sQ  [16][HD + 1];     // pre-scaled q rows for this head
  __shared__ float sKV [16][HD + 1];     // k (pass1) / v (pass2) j-tile
  __shared__ float sE  [256][HD + 1];    // projected edge tile, rows p=ii*16+jj
  __shared__ float sSim[16][N_NODES + 1];// sim strip -> exp*adj weights
  __shared__ float sRed[8][16];
  __shared__ float sMax[16];
  __shared__ float sInv[16];

  // q tile, pre-multiplied by 1/sqrt(D)
  const float qscale = 0.17677669529663688f;  // 1/sqrt(32)
  for (int idx = tid; idx < 16*HD; idx += 128) {
    const int r = idx >> 5, d = idx & 31;
    sQ[r][d] = qbuf[(size_t)(i0 + r) * INNER + h*HD + d] * qscale;
  }

  // We fragments for this head: K=64 -> 2 ksteps, N=32 -> 2 n-tiles (constant).
  v16bf bw[2][2];
#pragma unroll
  for (int ks = 0; ks < 2; ++ks)
#pragma unroll
    for (int nt = 0; nt < 2; ++nt)
      bw[ks][nt] = load_b_frag(We, INNER, ks*32, h*HD + nt*16 + (lane & 15), g);

  // E(j0): project 16x16 (i,j) pair block: 256 rows x EDIM -> 256 x 32.
  // Wave w owns m-tiles w, w+4, w+8, w+12 (16 WMMAs per wave per call).
  auto computeE = [&](int j0) {
#pragma unroll
    for (int mi = 0; mi < 4; ++mi) {
      const int mt = wv + mi*4;
      const int p  = mt*16 + (lane & 15);          // pair row 0..255
      const int ii = p >> 4, jj = p & 15;
      const float* rowptr =
          edges + ((size_t)(i0 + ii) * N_NODES + (j0 + jj)) * EDIM;
      if (j0 + 16 < N_NODES)
        __builtin_prefetch(rowptr + 16 * EDIM, 0, 1);  // next j-tile, same (ii,jj)
      const v16bf a0 = load_a_frag(rowptr,      g);
      const v16bf a1 = load_a_frag(rowptr + 32, g);
#pragma unroll
      for (int nt = 0; nt < 2; ++nt) {
        v8f acc = {};
        acc = wmma_bf16(a0, bw[0][nt], acc);
        acc = wmma_bf16(a1, bw[1][nt], acc);
        const int col = nt*16 + (lane & 15);
        const float bb = be[h*HD + col];
#pragma unroll
        for (int r = 0; r < 8; ++r)
          sE[mt*16 + r + 8*g][col] = acc[r] + bb;
      }
    }
  };

  // ---------------- Pass 1: sim strip ----------------
  for (int jt = 0; jt < N_NODES/16; ++jt) {
    const int j0 = jt * 16;
    computeE(j0);
    for (int idx = tid; idx < 16*HD; idx += 128) {
      const int jj = idx >> 5, d = idx & 31;
      sKV[jj][d] = kbuf[(size_t)(j0 + jj) * INNER + h*HD + d];
    }
    __syncthreads();
    for (int p = tid; p < 256; p += 128) {
      const int ii = p >> 4, jj = p & 15;
      float s = 0.f;
#pragma unroll
      for (int d = 0; d < HD; ++d)
        s += sQ[ii][d] * (sKV[jj][d] + sE[p][d]);
      sSim[ii][j0 + jj] = s;
    }
    __syncthreads();
  }

  // ---------------- masked softmax (max-subtraction == ref's mean quirk) ---
  {
    const int row = tid & 15, chunk = tid >> 4;    // 8 chunks of 64 cols
    float m = -1e30f;
    for (int j = chunk*64; j < chunk*64 + 64; ++j) m = fmaxf(m, sSim[row][j]);
    sRed[chunk][row] = m;
  }
  __syncthreads();
  if (tid < 16) {
    float m = sRed[0][tid];
#pragma unroll
    for (int c = 1; c < 8; ++c) m = fmaxf(m, sRed[c][tid]);
    sMax[tid] = m;
  }
  __syncthreads();
  {
    const int row = tid & 15, chunk = tid >> 4;
    const float m = sMax[row];
    const int* arow = adj + (size_t)(i0 + row) * N_NODES;
    float s = 0.f;
    for (int j = chunk*64; j < chunk*64 + 64; ++j) {
      const float e = __expf(sSim[row][j] - m) * (float)arow[j];
      sSim[row][j] = e;
      s += e;
    }
    sRed[chunk][row] = s;
  }
  __syncthreads();
  if (tid < 16) {
    float s = 0.f;
#pragma unroll
    for (int c = 0; c < 8; ++c) s += sRed[c][tid];
    sInv[tid] = (s == 0.f) ? 1.f : 1.f / s;        // ref: x_sum==0 -> 1.0
  }
  __syncthreads();

  // ---------------- Pass 2: out = attn . (v + e) ----------------
  float acc[4] = {0.f, 0.f, 0.f, 0.f};
  const int oii = tid >> 3;            // output row 0..15
  const int od0 = (tid & 7) * 4;       // 4 consecutive d's
  for (int jt = 0; jt < N_NODES/16; ++jt) {
    const int j0 = jt * 16;
    computeE(j0);
    for (int idx = tid; idx < 16*HD; idx += 128) {
      const int jj = idx >> 5, d = idx & 31;
      sKV[jj][d] = vbuf[(size_t)(j0 + jj) * INNER + h*HD + d];
    }
    __syncthreads();
#pragma unroll
    for (int jj = 0; jj < 16; ++jj) {
      const float aw = sSim[oii][j0 + jj];
#pragma unroll
      for (int u = 0; u < 4; ++u)
        acc[u] += aw * (sKV[jj][od0 + u] + sE[oii*16 + jj][od0 + u]);
    }
    __syncthreads();
  }
  const float inv = sInv[oii];
#pragma unroll
  for (int u = 0; u < 4; ++u)
    out[(size_t)(i0 + oii) * INNER + h*HD + od0 + u] = acc[u] * inv;
}

// ---------------------------------------------------------------------------
extern "C" void kernel_launch(void* const* d_in, const int* in_sizes, int n_in,
                              void* d_out, int out_size, void* d_ws, size_t ws_size,
                              hipStream_t stream) {
  const float* nodes = (const float*)d_in[0];
  const float* edges = (const float*)d_in[1];
  const int*   adjacency = (const int*)d_in[2];
  const float* Wq = (const float*)d_in[3];
  const float* bq = (const float*)d_in[4];
  const float* Wk = (const float*)d_in[5];
  const float* bk = (const float*)d_in[6];
  const float* Wv = (const float*)d_in[7];
  const float* bv = (const float*)d_in[8];
  const float* We = (const float*)d_in[9];
  const float* be = (const float*)d_in[10];
  float* out = (float*)d_out;

  float* qbuf = (float*)d_ws;                 // 512*256 f32
  float* kbuf = qbuf + N_NODES * INNER;
  float* vbuf = kbuf + N_NODES * INNER;

  dim3 gproj(N_NODES / 16, 3);
  gt_qkv_proj<<<gproj, 128, 0, stream>>>(nodes, Wq, bq, Wk, bk, Wv, bv,
                                         qbuf, kbuf, vbuf);

  dim3 gattn(N_NODES / 16, HEADS);
  gt_attn<<<gattn, 128, 0, stream>>>(edges, adjacency, We, be,
                                     qbuf, kbuf, vbuf, out);
}
